// ShiftedWindowMSA_77945066488323
// MI455X (gfx1250) — matrix operand order repaired and checked by reading, not compiled, not run on hardware
//
#include <hip/hip_runtime.h>
#include <hip/hip_bf16.h>
#include <stdint.h>

typedef __attribute__((ext_vector_type(16))) __bf16 v16bf;
typedef __attribute__((ext_vector_type(8)))  float  v8f;

union BF16x16 { v16bf v; unsigned short h[16]; uint32_t d[8]; };

__device__ __forceinline__ unsigned short f2bf(float f) {
  union { float f; uint32_t u; } x; x.f = f;
  uint32_t u = x.u;
  u += 0x7fffu + ((u >> 16) & 1u);         // round-to-nearest-even
  return (unsigned short)(u >> 16);
}

// problem dims
constexpr int NE = 256;     // embed
constexpr int NQ = 768;     // qkv columns
constexpr int WS = 49;      // window 7x7
constexpr int MP = 64;      // padded window rows

// LDS layout (bytes), phase-aliased
constexpr int QS2  = 40;    // q_s / v_s row stride (bf16): [h][m(64)][d(32)]
constexpr int KS   = 72;    // k_s row stride (bf16):      [h][d(32)][key(64)]
constexpr int ASTR = 264;   // bf16 x-tile stride (ushort)
constexpr int SSTR = 68;    // scores stride (float); P row = 136 ushort = 272B (16B aligned)

constexpr int Q_BYTES  = 8 * 64 * QS2 * 2;   // 40960
constexpr int K_BYTES  = 8 * 32 * KS  * 2;   // 36864
constexpr int V_BYTES  = Q_BYTES;            // 40960
constexpr int R2_OFF   = Q_BYTES + K_BYTES + V_BYTES;       // 118784
constexpr int A_BYTES  = MP * ASTR * 2;      // 33792
constexpr int ST_BYTES = WS * NE * 4;        // 50176 f32 staging
constexpr int SC_BYTES = 8 * WS * SSTR * 4;  // 106624
constexpr int R2_BYTES = (A_BYTES + ST_BYTES) > SC_BYTES ? (A_BYTES + ST_BYTES) : SC_BYTES;
constexpr int SMEM_BYTES = R2_OFF + R2_BYTES;               // 225408 (+ pos table)

__global__ void cvt_w1_bf16(const float* __restrict__ w, unsigned short* __restrict__ o, int n) {
  int i = blockIdx.x * blockDim.x + threadIdx.x;
  if (i < n) o[i] = f2bf(w[i]);
}

__global__ __launch_bounds__(256, 1)
void swin_msa_fused(const float* __restrict__ x, const unsigned short* __restrict__ w1b,
                    const float* __restrict__ b1, const float* __restrict__ pos,
                    float* __restrict__ out)
{
  __shared__ __align__(16) unsigned char smem[SMEM_BYTES];
  __shared__ float pos_s[13 * 13];
  unsigned short* q_s  = (unsigned short*)smem;                       // [8][64][QS2]
  unsigned short* k_s  = (unsigned short*)(smem + Q_BYTES);           // [8][32][KS]
  unsigned short* v_s  = (unsigned short*)(smem + Q_BYTES + K_BYTES); // [8][64][QS2]
  unsigned short* a_s  = (unsigned short*)(smem + R2_OFF);            // x tile bf16 (phase 1)
  float*          stg  = (float*)(smem + R2_OFF + A_BYTES);           // x tile f32 staging
  float*          sc_s = (float*)(smem + R2_OFF);                     // scores (phase 2)

  const int tid  = threadIdx.x;
  const int lane = tid & 31;
  const int wave = tid >> 5;
  const int hlf  = lane >> 4;
  const int l16  = lane & 15;

  const int blk = blockIdx.x;
  const int v   = blk & 7;          // window col
  const int u   = (blk >> 3) & 7;   // window row
  const int b   = blk >> 6;         // batch

  const int nbase = (wave >> 2) * 24;   // phase-1 n-tile base for this wave

  // warm WGP caches for this wave's first B tiles (global_prefetch_b8)
#pragma unroll
  for (int kt = 0; kt < 8; ++kt)
    __builtin_prefetch(&w1b[(size_t)(kt * 32 + lane) * NQ + nbase * 16], 0, 3);

  // stage relative-position bias table into LDS
  for (int i = tid; i < 169; i += 256) pos_s[i] = pos[i];

  // ---- async-copy rolled x window (f32) into LDS staging: 49 x 64 x b128 ----
  for (int i = tid; i < WS * 64; i += 256) {
    int m = i >> 6, seg = i & 63;
    int wr = m / 7, wc = m - wr * 7;
    int sh = (u * 7 + wr + 3) % 56;
    int sw = (v * 7 + wc + 3) % 56;
    uint64_t gaddr = (uint64_t)(uintptr_t)&x[(((size_t)b * 56 + sh) * 56 + sw) * 256 + seg * 4];
    uint32_t laddr = (uint32_t)(uintptr_t)&stg[i * 4];   // low 32 bits = LDS offset
    asm volatile("global_load_async_to_lds_b128 %0, %1, off"
                 :: "v"(laddr), "v"(gaddr) : "memory");
  }
  asm volatile("s_wait_asynccnt 0x0" ::: "memory");
  __syncthreads();

  // ---- convert staged f32 -> bf16 A tile; zero-pad rows 49..63 ----
  for (int i = tid; i < WS * 64; i += 256) {
    float4 f = ((const float4*)stg)[i];
    uint2 pk;
    pk.x = (uint32_t)f2bf(f.x) | ((uint32_t)f2bf(f.y) << 16);
    pk.y = (uint32_t)f2bf(f.z) | ((uint32_t)f2bf(f.w) << 16);
    int m = i >> 6, seg = i & 63;
    *(uint2*)&a_s[m * ASTR + seg * 4] = pk;
  }
  for (int i = WS * NE + tid; i < MP * NE; i += 256)
    a_s[(i >> 8) * ASTR + (i & 255)] = 0;
  __syncthreads();

  // ---- phase 1: qkv = A[64x256] @ w1b[256x768] + b1, scattered to q/k/v head-major ----
  {
    const int mtile = wave & 3;
    BF16x16 af[8];
    const int am = mtile * 16 + l16;
#pragma unroll
    for (int kt = 0; kt < 8; ++kt) {
#pragma unroll
      for (int r = 0; r < 8; ++r) {
        int kk = kt * 32 + (r < 4 ? hlf * 8 + 2 * r : 16 + hlf * 8 + 2 * (r - 4));
        af[kt].d[r] = *(const uint32_t*)&a_s[am * ASTR + kk];
      }
    }
    for (int nt = 0; nt < 24; ++nt) {
      const int n0 = (nbase + nt) * 16;
      v8f acc = {0.f, 0.f, 0.f, 0.f, 0.f, 0.f, 0.f, 0.f};
#pragma unroll
      for (int kt = 0; kt < 8; ++kt) {
        BF16x16 bf;                          // lane = N column, 16 K values (table order)
#pragma unroll
        for (int s = 0; s < 16; ++s) {
          int kk = kt * 32 + (s < 8 ? hlf * 8 + s : 16 + hlf * 8 + (s - 8));
          bf.h[s] = w1b[(size_t)kk * NQ + n0 + l16];
        }
        acc = __builtin_amdgcn_wmma_f32_16x16x32_bf16(false, af[kt].v, false, bf.v,
                                                      (short)0, acc, false, false);
      }
      // epilogue: n = 3*e + c, e = d*8 + head -> scatter into q/k/v fragment layouts
      const int n  = n0 + l16;
      const int e  = n / 3, c = n - e * 3;
      const int hh = e & 7, dd = e >> 3;
      const float bb = b1[n];
      unsigned short* dst;
      int inc;
      if (c == 1) { dst = &k_s[(hh * 32 + dd) * KS + mtile * 16 + hlf * 8]; inc = 1; }
      else {
        unsigned short* base = (c == 0) ? q_s : v_s;
        dst = &base[(size_t)(hh * 64 + mtile * 16 + hlf * 8) * QS2 + dd]; inc = QS2;
      }
#pragma unroll
      for (int r = 0; r < 8; ++r) dst[r * inc] = f2bf(acc[r] + bb);
    }
  }
  __syncthreads();

  // ---- phase 2: attention, wave == head ----
  const int h = wave;
  float* sch = &sc_s[h * WS * SSTR];

  { // S = q @ k^T : q fragments are pairs of ds_load_b128, k built per element
    BF16x16 qf[4];
#pragma unroll
    for (int mt = 0; mt < 4; ++mt) {
      const unsigned short* qrow = &q_s[(size_t)(h * 64 + mt * 16 + l16) * QS2];
      *(uint4*)&qf[mt].d[0] = *(const uint4*)&qrow[hlf * 8];
      *(uint4*)&qf[mt].d[4] = *(const uint4*)&qrow[16 + hlf * 8];
    }
#pragma unroll
    for (int nt = 0; nt < 4; ++nt) {
      BF16x16 kf;                            // lane = key column, 16 dims (table order)
#pragma unroll
      for (int s = 0; s < 16; ++s) {
        int kk = s < 8 ? hlf * 8 + s : 16 + hlf * 8 + (s - 8);
        kf.h[s] = k_s[(size_t)(h * 32 + kk) * KS + nt * 16 + l16];
      }
#pragma unroll
      for (int mt = 0; mt < 4; ++mt) {
        v8f acc = {0.f, 0.f, 0.f, 0.f, 0.f, 0.f, 0.f, 0.f};
        acc = __builtin_amdgcn_wmma_f32_16x16x32_bf16(false, qf[mt].v, false, kf.v,
                                                      (short)0, acc, false, false);
#pragma unroll
        for (int r = 0; r < 8; ++r) {
          int m = mt * 16 + hlf * 8 + r;
          if (m < WS) sch[m * SSTR + nt * 16 + l16] = acc[r];
        }
      }
    }
  }
  __syncthreads();

  { // scale + bias + shift masks + softmax; write P (bf16) in place over scores
    const float scale = 0.17677669529663687f;    // 1/sqrt(32)
    const bool lastu = (u == 7), lastv = (v == 7);
    for (int m = lane; m < WS; m += 32) {
      float* srow = &sch[m * SSTR];
      int q7 = m / 7, qm = m - q7 * 7;
      float mx = -3.4e38f;
      int k7 = 0, km = 0;
      for (int k = 0; k < WS; ++k) {
        float s = srow[k] * scale + pos_s[(k7 - q7 + 6) * 13 + (km - qm + 6)];
        if (lastu && ((m >= 28) != (k >= 28))) s = -1e30f;
        if (lastv && ((qm >= 4) != (km >= 4))) s = -1e30f;
        srow[k] = s;
        mx = fmaxf(mx, s);
        if (++km == 7) { km = 0; ++k7; }
      }
      float sum = 0.f;
      for (int k = 0; k < WS; ++k) {
        float e = __expf(srow[k] - mx);
        srow[k] = e;
        sum += e;
      }
      float rs = 1.f / sum;
      unsigned short* prow = (unsigned short*)srow;   // in-place narrowing is overlap-safe
      for (int k = 0; k < WS; ++k) prow[k] = f2bf(srow[k] * rs);
      for (int k = WS; k < 64; ++k) prow[k] = 0;      // zero padded keys
    }
  }
  __syncthreads();

  { // O = P @ V, rolled scatter store
    const unsigned short* pbase = (const unsigned short*)sch;  // P stride = 136 ushort (272B)
#pragma unroll
    for (int mt = 0; mt < 4; ++mt) {
      int m  = mt * 16 + l16;
      int mr = m < WS ? m : WS - 1;          // clamp padded rows
      BF16x16 pf[2];
#pragma unroll
      for (int kt = 0; kt < 2; ++kt) {
        const unsigned short* prow = &pbase[mr * (SSTR * 2) + kt * 32];
        *(uint4*)&pf[kt].d[0] = *(const uint4*)&prow[hlf * 8];
        *(uint4*)&pf[kt].d[4] = *(const uint4*)&prow[16 + hlf * 8];
      }
#pragma unroll
      for (int dt = 0; dt < 2; ++dt) {
        v8f acc = {0.f, 0.f, 0.f, 0.f, 0.f, 0.f, 0.f, 0.f};
#pragma unroll
        for (int kt = 0; kt < 2; ++kt) {
          BF16x16 vf;                        // lane = dim column, 16 keys (table order)
#pragma unroll
          for (int s = 0; s < 16; ++s) {
            int kk = s < 8 ? hlf * 8 + s : 16 + hlf * 8 + (s - 8);
            vf.h[s] = v_s[(size_t)(h * 64 + kt * 32 + kk) * QS2 + dt * 16 + l16];
          }
          acc = __builtin_amdgcn_wmma_f32_16x16x32_bf16(false, pf[kt].v, false, vf.v,
                                                        (short)0, acc, false, false);
        }
#pragma unroll
        for (int r = 0; r < 8; ++r) {
          int mm = mt * 16 + hlf * 8 + r;
          if (mm < WS) {
            int wr = mm / 7, wc = mm - wr * 7;
            int dh = (u * 7 + wr + 3) % 56;
            int dw = (v * 7 + wc + 3) % 56;
            out[(((size_t)b * 56 + dh) * 56 + dw) * 256 + (dt * 16 + l16) * 8 + h] = acc[r];
          }
        }
      }
    }
  }
}

extern "C" void kernel_launch(void* const* d_in, const int* in_sizes, int n_in,
                              void* d_out, int out_size, void* d_ws, size_t ws_size,
                              hipStream_t stream) {
  (void)in_sizes; (void)n_in; (void)out_size; (void)ws_size;
  const float* x   = (const float*)d_in[0];
  const float* w1  = (const float*)d_in[1];
  const float* b1  = (const float*)d_in[2];
  const float* pos = (const float*)d_in[3];
  float* out = (float*)d_out;
  unsigned short* w1b = (unsigned short*)d_ws;   // 256*768 bf16 = 384 KB scratch

  int nw = NE * NQ;
  cvt_w1_bf16<<<(nw + 255) / 256, 256, 0, stream>>>(w1, w1b, nw);
  swin_msa_fused<<<32 * 8 * 8, 256, 0, stream>>>(x, w1b, b1, pos, out);
}
